// SelfAttention_11656541241579
// MI455X (gfx1250) — matrix operand order repaired and checked by reading, not compiled, not run on hardware
//
#include <hip/hip_runtime.h>
#include <hip/hip_bf16.h>

// ---------------------------------------------------------------------------
// CDNA5 (gfx1250) fused GroupNorm + MHSA + proj block, bf16 WMMA pipeline.
// All hot fragment loads are contiguous 16B (b128) accesses; attention stages
// K/V tiles in LDS cooperatively (flash-attention style, wave32 WMMA).
// ---------------------------------------------------------------------------

typedef __bf16 v16bf __attribute__((ext_vector_type(16)));
typedef __bf16 v8bf  __attribute__((ext_vector_type(8)));
typedef float  v8f   __attribute__((ext_vector_type(8)));

#define CCH   256          // channels
#define NSP   1024         // H*W
#define BATCH 16
#define HEADS 4
#define DH    64           // CCH / HEADS
#define GRP   8

static __device__ __forceinline__ unsigned short f2bf(float f) {
    unsigned u = __float_as_uint(f);
    unsigned r = (u + 0x7FFFu + ((u >> 16) & 1u)) >> 16;   // RNE
    return (unsigned short)r;
}
static __device__ __forceinline__ v8f vzero8() {
    v8f c;
#pragma unroll
    for (int j = 0; j < 8; ++j) c[j] = 0.0f;
    return c;
}
static __device__ __forceinline__ v8f wmma_bf16(v16bf a, v16bf b, v8f c) {
    return __builtin_amdgcn_wmma_f32_16x16x32_bf16(false, a, false, b,
                                                   (short)0, c, false, false);
}

// 8 contiguous bf16 via one 16-byte load
static __device__ __forceinline__ v8bf ld8bf(const unsigned short* p) {
    union { uint4 u; v8bf b; } x;
    x.u = *(const uint4*)p;
    return x.b;
}
// 16 contiguous bf16 via two 16-byte loads
static __device__ __forceinline__ v16bf ld16bf(const unsigned short* p) {
    return __builtin_shufflevector(ld8bf(p), ld8bf(p + 8),
        0, 1, 2, 3, 4, 5, 6, 7, 8, 9, 10, 11, 12, 13, 14, 15);
}
// A-fragment 16x32 bf16: rowp points at row start (+k0). Per ISA layout:
// elems 0..7 -> K = 8*hl + e ; elems 8..15 -> K = 16 + 8*hl + e.
static __device__ __forceinline__ v16bf load_a_frag(const unsigned short* rowp,
                                                    int lane) {
    int kb = 8 * (lane >> 4);
    return __builtin_shufflevector(ld8bf(rowp + kb), ld8bf(rowp + kb + 16),
        0, 1, 2, 3, 4, 5, 6, 7, 8, 9, 10, 11, 12, 13, 14, 15);
}
// B-fragment 32x16 bf16: colp points at this lane's column (+k0);
// elems 0..7 -> K = 8*hl + e ; elems 8..15 -> K = 16 + 8*hl + e.
static __device__ __forceinline__ v16bf load_b_frag(const unsigned short* colp,
                                                    int lane) {
    int kb = 8 * (lane >> 4);
    return __builtin_shufflevector(ld8bf(colp + kb), ld8bf(colp + kb + 16),
        0, 1, 2, 3, 4, 5, 6, 7, 8, 9, 10, 11, 12, 13, 14, 15);
}

static __device__ __forceinline__ float half16_max(float v) {
    v = fmaxf(v, __shfl_xor(v, 1, 32));
    v = fmaxf(v, __shfl_xor(v, 2, 32));
    v = fmaxf(v, __shfl_xor(v, 4, 32));
    v = fmaxf(v, __shfl_xor(v, 8, 32));
    return v;
}
static __device__ __forceinline__ float half16_sum(float v) {
    v += __shfl_xor(v, 1, 32);
    v += __shfl_xor(v, 2, 32);
    v += __shfl_xor(v, 4, 32);
    v += __shfl_xor(v, 8, 32);
    return v;
}

// ---------------------------------------------------------------------------
// 1) fp32 -> bf16 weight conversion (qkv_w: 3C*C, proj_w: C*C)
// ---------------------------------------------------------------------------
__global__ void cvt_w_kernel(const float* __restrict__ qkvw,
                             const float* __restrict__ projw,
                             unsigned short* __restrict__ qkvw_bf,
                             unsigned short* __restrict__ projw_bf) {
    int i = blockIdx.x * blockDim.x + threadIdx.x;
    if (i < 3 * CCH * CCH) qkvw_bf[i] = f2bf(qkvw[i]);
    else                   projw_bf[i - 3 * CCH * CCH] = f2bf(projw[i - 3 * CCH * CCH]);
}

// ---------------------------------------------------------------------------
// 2) GroupNorm: one block per (b, group); bf16 output TRANSPOSED xnT[B][N][C]
// ---------------------------------------------------------------------------
__global__ void gn_kernel(const float* __restrict__ x,
                          const float* __restrict__ w,
                          const float* __restrict__ bia,
                          unsigned short* __restrict__ xnT) {
    const int tid = threadIdx.x;
    const int bg  = blockIdx.x;
    const int bb  = bg >> 3, g = bg & 7;
    const int ELEMS = (CCH / GRP) * NSP;   // 32768
    const float* src = x + ((size_t)bb * CCH + g * (CCH / GRP)) * NSP;
    unsigned short* dst = xnT + (size_t)bb * NSP * CCH;

    float s = 0.0f, s2 = 0.0f;
    for (int i = tid; i < ELEMS; i += 256) {
        float v = src[i];
        s += v; s2 += v * v;
    }
    __shared__ float red0[256], red1[256];
    red0[tid] = s; red1[tid] = s2;
    __syncthreads();
    for (int off = 128; off > 0; off >>= 1) {
        if (tid < off) { red0[tid] += red0[tid + off]; red1[tid] += red1[tid + off]; }
        __syncthreads();
    }
    float mu   = red0[0] * (1.0f / ELEMS);
    float var  = red1[0] * (1.0f / ELEMS) - mu * mu;
    float rstd = rsqrtf(var + 1e-5f);
    for (int i = tid; i < ELEMS; i += 256) {
        int cl = i >> 10, n = i & (NSP - 1);
        int c  = g * (CCH / GRP) + cl;
        float v = (src[i] - mu) * rstd * w[c] + bia[c];
        dst[(size_t)n * CCH + c] = f2bf(v);
    }
}

// ---------------------------------------------------------------------------
// 3) QKV GEMM: Wqkv(768x256) * Xn[b](256x1024) + bias.
//    Q/K rows (o<512) -> qkT[B][N][512] (one b128 store per lane),
//    V rows (o>=512)  -> vbuf[B][256][N] (d-major = PV B-operand layout).
//    grid (64, 6, 16), block 256 (8 waves -> 8 m-tiles)
// ---------------------------------------------------------------------------
__global__ void qkv_gemm_kernel(const unsigned short* __restrict__ wbf,
                                const unsigned short* __restrict__ xnT,
                                const float* __restrict__ bias,
                                unsigned short* __restrict__ qkT,
                                unsigned short* __restrict__ vbuf) {
    const int lane = threadIdx.x & 31;
    const int wid  = threadIdx.x >> 5;
    const int hl   = lane >> 4, ll = lane & 15;
    const int mbase = (blockIdx.y * 8 + wid) * 16;
    const int nbase = blockIdx.x * 16;
    const int b = blockIdx.z;
    const unsigned short* xb = xnT + (size_t)b * NSP * CCH;
    const int arow = mbase + ll;
    const int col  = nbase + ll;

    v8f c = vzero8();
#pragma unroll
    for (int kk = 0; kk < CCH; kk += 32) {
        v16bf a  = load_a_frag(wbf + (size_t)arow * CCH + kk, lane);
        v16bf bm = load_b_frag(xb + (size_t)col * CCH + kk, lane);
        c = wmma_bf16(a, bm, c);
    }
    if (mbase < 2 * CCH) {      // Q or K rows -> transposed, packed store
        union { unsigned short s[8]; uint4 u; } pk;
#pragma unroll
        for (int j = 0; j < 8; ++j) {
            int row = mbase + j + 8 * hl;
            pk.s[j] = f2bf(c[j] + bias[row]);
        }
        unsigned short* dst = qkT + (size_t)b * NSP * (2 * CCH)
                            + (size_t)col * (2 * CCH) + mbase + 8 * hl;
        *(uint4*)dst = pk.u;
    } else {                    // V rows -> d-major [256][N]
        unsigned short* vb = vbuf + (size_t)b * CCH * NSP;
#pragma unroll
        for (int j = 0; j < 8; ++j) {
            int row = mbase + j + 8 * hl;
            vb[(size_t)(row - 2 * CCH) * NSP + col] = f2bf(c[j] + bias[row]);
        }
    }
}

// ---------------------------------------------------------------------------
// 4) Fused flash attention: grid (8, 64) block 256; wave owns 16 query rows;
//    block cooperatively stages the shared 32-wide K/V tile in LDS.
// ---------------------------------------------------------------------------
__global__ void attn_kernel(const unsigned short* __restrict__ qkT,
                            const unsigned short* __restrict__ vbuf,
                            unsigned short* __restrict__ attnT) {
    __shared__ unsigned short kv_k[32][64];    // [m][d] K tile   (4 KB)
    __shared__ unsigned short kv_v[64][32];    // [d][m] V tile   (4 KB)
    __shared__ unsigned short lds_p[8][16][32]; // per-wave P tile (8 KB)

    const int tid  = threadIdx.x;
    const int lane = tid & 31;
    const int wid  = tid >> 5;
    const int hl = lane >> 4, ll = lane & 15;
    const int bh = blockIdx.y;
    const int b = bh >> 2, h = bh & 3;
    const int qrow0 = (blockIdx.x * 8 + wid) * 16;
    const unsigned short* qkb = qkT + (size_t)b * NSP * (2 * CCH); // [n][512]
    const int qoff = h * DH;
    const int koff = CCH + h * DH;
    const unsigned short* vb = vbuf + (size_t)b * CCH * NSP + (size_t)(h * DH) * NSP;
    const float scale = 0.125f;   // 1/sqrt(64)

    // staging indices (each thread moves one 16B chunk of K and one of V)
    const int kml = tid >> 3, kdo = (tid & 7) * 8;   // K: m row, d offset
    const int vdl = tid >> 2, vmo = (tid & 3) * 8;   // V: d row, m offset

    // Q A-fragments (K-dim = d, contiguous in qkT rows), live in registers
    v16bf aq0 = load_a_frag(qkb + (size_t)(qrow0 + ll) * (2 * CCH) + qoff, lane);
    v16bf aq1 = load_a_frag(qkb + (size_t)(qrow0 + ll) * (2 * CCH) + qoff + 32, lane);

    float mst[8], lst[8];
    v8f oacc[4];
#pragma unroll
    for (int j = 0; j < 8; ++j) { mst[j] = -1e30f; lst[j] = 0.0f; }
#pragma unroll
    for (int t = 0; t < 4; ++t) oacc[t] = vzero8();

    for (int mb = 0; mb < NSP; mb += 32) {
        __syncthreads();   // previous iteration's LDS consumers are done
        *(uint4*)&kv_k[kml][kdo] =
            *(const uint4*)(qkb + (size_t)(mb + kml) * (2 * CCH) + koff + kdo);
        *(uint4*)&kv_v[vdl][vmo] =
            *(const uint4*)(vb + (size_t)vdl * NSP + mb + vmo);
        __syncthreads();

        // S = scale * Q K^T, two 16-col subtiles from LDS (contiguous reads)
        v8f s0 = vzero8(), s1 = vzero8();
        s0 = wmma_bf16(aq0, load_b_frag(&kv_k[ll][0],       lane), s0);
        s0 = wmma_bf16(aq1, load_b_frag(&kv_k[ll][32],      lane), s0);
        s1 = wmma_bf16(aq0, load_b_frag(&kv_k[16 + ll][0],  lane), s1);
        s1 = wmma_bf16(aq1, load_b_frag(&kv_k[16 + ll][32], lane), s1);

        float alpha[8];
#pragma unroll
        for (int j = 0; j < 8; ++j) {
            s0[j] *= scale; s1[j] *= scale;
            float bmax = half16_max(fmaxf(s0[j], s1[j]));
            float mnew = fmaxf(mst[j], bmax);
            alpha[j] = expf(mst[j] - mnew);
            s0[j] = expf(s0[j] - mnew);
            s1[j] = expf(s1[j] - mnew);
            float rs = half16_sum(s0[j] + s1[j]);
            lst[j] = alpha[j] * lst[j] + rs;
            mst[j] = mnew;
        }
#pragma unroll
        for (int t = 0; t < 4; ++t)
#pragma unroll
            for (int j = 0; j < 8; ++j) oacc[t][j] *= alpha[j];

        // P: C-layout -> A-layout via per-wave LDS tile
#pragma unroll
        for (int j = 0; j < 8; ++j) {
            lds_p[wid][j + 8 * hl][ll]      = f2bf(s0[j]);
            lds_p[wid][j + 8 * hl][16 + ll] = f2bf(s1[j]);
        }
        __syncthreads();

        v16bf ap = load_a_frag(&lds_p[wid][ll][0], lane);
#pragma unroll
        for (int t = 0; t < 4; ++t)
            oacc[t] = wmma_bf16(ap, load_b_frag(&kv_v[t * 16 + ll][0], lane),
                                oacc[t]);
    }

    // normalize and store transposed: attnT[b][n][h*DH + d]
    unsigned short* ob = attnT + (size_t)b * NSP * CCH + h * DH;
#pragma unroll
    for (int t = 0; t < 4; ++t) {
        int d = t * 16 + ll;
#pragma unroll
        for (int j = 0; j < 8; ++j) {
            int n = qrow0 + j + 8 * hl;
            ob[(size_t)n * CCH + d] = f2bf(oacc[t][j] / lst[j]);
        }
    }
}

// ---------------------------------------------------------------------------
// 5) Proj GEMM + bias + residual: out = Wp(256x256) * A[b] + b + x
//    grid (64, 2, 16), block 256
// ---------------------------------------------------------------------------
__global__ void proj_kernel(const unsigned short* __restrict__ wbf,
                            const unsigned short* __restrict__ attnT,
                            const float* __restrict__ bias,
                            const float* __restrict__ x,
                            float* __restrict__ out) {
    const int lane = threadIdx.x & 31;
    const int wid  = threadIdx.x >> 5;
    const int hl = lane >> 4, ll = lane & 15;
    const int mbase = (blockIdx.y * 8 + wid) * 16;
    const int nbase = blockIdx.x * 16;
    const int b = blockIdx.z;
    const unsigned short* ab = attnT + (size_t)b * NSP * CCH;
    const int arow = mbase + ll;
    const int col  = nbase + ll;

    v8f c = vzero8();
#pragma unroll
    for (int kk = 0; kk < CCH; kk += 32) {
        v16bf a  = load_a_frag(wbf + (size_t)arow * CCH + kk, lane);
        v16bf bm = load_b_frag(ab + (size_t)col * CCH + kk, lane);
        c = wmma_bf16(a, bm, c);
    }
#pragma unroll
    for (int j = 0; j < 8; ++j) {
        int row = mbase + j + 8 * hl;
        size_t idx = (size_t)b * CCH * NSP + (size_t)row * NSP + col;
        out[idx] = c[j] + bias[row] + x[idx];
    }
}

// ---------------------------------------------------------------------------
extern "C" void kernel_launch(void* const* d_in, const int* in_sizes, int n_in,
                              void* d_out, int out_size, void* d_ws, size_t ws_size,
                              hipStream_t stream) {
    (void)in_sizes; (void)n_in; (void)out_size; (void)ws_size;
    const float* x     = (const float*)d_in[0];
    const float* gn_w  = (const float*)d_in[1];
    const float* gn_b  = (const float*)d_in[2];
    const float* qkv_w = (const float*)d_in[3];
    const float* qkv_b = (const float*)d_in[4];
    const float* pr_w  = (const float*)d_in[5];
    const float* pr_b  = (const float*)d_in[6];
    float* out = (float*)d_out;

    // workspace layout (bf16 as ushort), total ~42 MB
    unsigned short* ws       = (unsigned short*)d_ws;
    unsigned short* xnT      = ws;                                     // B*N*C
    unsigned short* qkT      = xnT  + (size_t)BATCH * NSP * CCH;       // B*N*512
    unsigned short* vbuf     = qkT  + (size_t)BATCH * NSP * 2 * CCH;   // B*C*N
    unsigned short* attnT    = vbuf + (size_t)BATCH * CCH * NSP;       // B*N*C
    unsigned short* qkvw_bf  = attnT + (size_t)BATCH * NSP * CCH;      // 3C*C
    unsigned short* projw_bf = qkvw_bf + (size_t)3 * CCH * CCH;        // C*C

    cvt_w_kernel<<<dim3((4 * CCH * CCH) / 256), 256, 0, stream>>>(
        qkv_w, pr_w, qkvw_bf, projw_bf);
    gn_kernel<<<dim3(BATCH * GRP), 256, 0, stream>>>(x, gn_w, gn_b, xnT);
    qkv_gemm_kernel<<<dim3(NSP / 16, (3 * CCH / 16) / 8, BATCH), 256, 0, stream>>>(
        qkvw_bf, xnT, qkv_b, qkT, vbuf);
    attn_kernel<<<dim3((NSP / 16) / 8, BATCH * HEADS), 256, 0, stream>>>(
        qkT, vbuf, attnT);
    proj_kernel<<<dim3(NSP / 16, (CCH / 16) / 8, BATCH), 256, 0, stream>>>(
        projw_bf, attnT, pr_b, x, out);
}
